// LinearAttentionClassifier_64733747085412
// MI455X (gfx1250) — compile-verified, matrix operand order checked by reading, not run on hardware
//
#include <hip/hip_runtime.h>
#include <hip/hip_bf16.h>
#include <math.h>

typedef _Float16 v16h __attribute__((ext_vector_type(16)));
typedef _Float16 v8h  __attribute__((ext_vector_type(8)));
typedef float    v8f  __attribute__((ext_vector_type(8)));

#define DIM  64
#define SEQ  784
#define TB   16
#define NBLK 49   // 49 * 16 = 784

// Build a v16h WMMA fragment (A or B) from an LDS row in the gfx1250 16-bit
// layout: lane L (hs = L>>4) holds K = chunk*32 + {hs*8..hs*8+7} in halves 0..7
// and K = chunk*32 + 16 + {hs*8..hs*8+7} in halves 8..15.
__device__ __forceinline__ v16h ldfrag(const _Float16* rowbase, int hs, int chunk) {
    const v8h lo = *(const v8h*)(rowbase + chunk * 32 + hs * 8);
    const v8h hi = *(const v8h*)(rowbase + chunk * 32 + 16 + hs * 8);
    return __builtin_shufflevector(lo, hi, 0,1,2,3,4,5,6,7,8,9,10,11,12,13,14,15);
}

__global__ __launch_bounds__(64) void efla_scan_kernel(
    const float* __restrict__ x,      // (B, S, 1)
    const float* __restrict__ W_in,   // (64, 1)
    const float* __restrict__ pos,    // (S, 64)
    const float* __restrict__ Wq,     // (64, 64)
    const float* __restrict__ Wk,     // (64, 64)
    const float* __restrict__ Wv,     // (64, 64)
    const float* __restrict__ Wb,     // (1, 64)
    const float* __restrict__ bb,     // (1,)
    const float* __restrict__ ln_g,   // (64,)
    const float* __restrict__ ln_b,   // (64,)
    const float* __restrict__ Wc,     // (10, 64)
    const float* __restrict__ bc,     // (10,)
    float* __restrict__ out)          // (B, 10)
{
    __shared__ __align__(16) _Float16 sWk[DIM * DIM];  // 8 KB, f16 row-major
    __shared__ __align__(16) _Float16 sWv[DIM * DIM];  // 8 KB
    __shared__ __align__(16) _Float16 sHh[TB * DIM];   // 2 KB, f16 h tile
    __shared__ __align__(16) float    sH [TB * DIM];   // 4 KB, f32 h tile
    __shared__ __align__(16) float    sK [TB * DIM];   // 4 KB, k tile (f32)
    __shared__ __align__(16) float    sV [TB * DIM];   // 4 KB, v tile (f32)
    __shared__ float sAlpha[TB];
    __shared__ float sWbv[DIM];
    __shared__ float sVec[DIM];
    __shared__ float sLN[DIM];

    const int tid  = threadIdx.x;        // 0..63
    const int b    = blockIdx.x;         // batch element
    const int lane = tid & 31;
    const int wav  = tid >> 5;           // wave id: 0 or 1 (n-halves)
    const int hs   = (lane >> 4) & 1;    // lane half-select
    const int mrow = lane & 15;          // fragment row / col index

    // Preload f16 weight copies + Wb into LDS.
    for (int i = tid; i < DIM * DIM; i += 64) {
        sWk[i] = (_Float16)Wk[i];
        sWv[i] = (_Float16)Wv[i];
    }
    sWbv[tid] = Wb[tid];
    const float win = W_in[tid];
    const float bbv = bb[0];

    // Per-thread state column S[:, tid], kept in VGPRs.
    float S[DIM];
#pragma unroll
    for (int i = 0; i < DIM; ++i) S[i] = 0.0f;

    __syncthreads();

    for (int blk = 0; blk < NBLK; ++blk) {
        const int t0 = blk * TB;

        // ---- Phase A: h tile = gelu(x*W_in^T + pos) (exact erf GELU) ----
#pragma unroll 4
        for (int j = 0; j < TB; ++j) {
            const int t = t0 + j;
            const float z = fmaf(x[b * SEQ + t], win, pos[t * DIM + tid]);
            const float h = 0.5f * z * (1.0f + erff(z * 0.70710678118654752f));
            sH [j * DIM + tid] = h;
            sHh[j * DIM + tid] = (_Float16)h;
            if (blk + 1 < NBLK)  // hide next block's pos_emb latency
                __builtin_prefetch(&pos[(t + TB) * DIM + tid], 0, 0);
        }
        __syncthreads();

        // ---- Phase B: K/V tiles via WMMA; wave handles n in [wav*32, wav*32+32) ----
#pragma unroll
        for (int ti = 0; ti < 2; ++ti) {
            const int n0 = wav * 32 + ti * 16;
            v8f ck = {};
            v8f cv = {};
#pragma unroll
            for (int c = 0; c < 2; ++c) {   // K-dim chunks of 32
                const v16h a  = ldfrag(&sHh[mrow * DIM], hs, c);
                const v16h bk = ldfrag(&sWk[(n0 + mrow) * DIM], hs, c);
                ck = __builtin_amdgcn_wmma_f32_16x16x32_f16(
                        false, a, false, bk, (short)0, ck, false, false);
                const v16h bv = ldfrag(&sWv[(n0 + mrow) * DIM], hs, c);
                cv = __builtin_amdgcn_wmma_f32_16x16x32_f16(
                        false, a, false, bv, (short)0, cv, false, false);
            }
            // D layout: VGPR r -> row (r + hs*8), col (n0 + mrow)
#pragma unroll
            for (int r = 0; r < 8; ++r) {
                const int trow = r + hs * 8;
                sK[trow * DIM + n0 + mrow] = ck[r];
                sV[trow * DIM + n0 + mrow] = cv[r];
            }
        }
        __syncthreads();

        // ---- Phase C: per-step gate alpha, computed once in parallel ----
        // alpha_t = -expm1(-beta_t * lam_t) / (lam_t + 1e-6),
        //   lam_t = clip(||k_t||^2, 1e-6), beta_t = sigmoid(h_t.Wb + bb)
        if (tid < TB) {
            float acc = bbv;
#pragma unroll
            for (int d = 0; d < DIM; ++d)
                acc = fmaf(sH[tid * DIM + d], sWbv[d], acc);
            const float beta = 1.0f / (1.0f + expf(-acc));

            float l0 = 0.f, l1 = 0.f, l2 = 0.f, l3 = 0.f;
#pragma unroll
            for (int d = 0; d < DIM; d += 4) {
                const float4 kc = *(const float4*)&sK[tid * DIM + d];
                l0 = fmaf(kc.x, kc.x, l0);
                l1 = fmaf(kc.y, kc.y, l1);
                l2 = fmaf(kc.z, kc.z, l2);
                l3 = fmaf(kc.w, kc.w, l3);
            }
            const float lam  = (l0 + l1) + (l2 + l3);
            const float lamc = fmaxf(lam, 1e-6f);
            sAlpha[tid] = -expm1f(-beta * lamc) / (lamc + 1e-6f);
        }
        __syncthreads();

        // ---- Phase D: sequential EFLA scan over the 16 steps ----
        for (int j = 0; j < TB; ++j) {
            // cache k_t in registers (broadcast LDS reads, single pass)
            float kv[DIM];
#pragma unroll
            for (int c = 0; c < 16; ++c) {
                const float4 kc = *(const float4*)&sK[j * DIM + c * 4];
                kv[c*4+0] = kc.x; kv[c*4+1] = kc.y;
                kv[c*4+2] = kc.z; kv[c*4+3] = kc.w;
            }
            // kS = k . S[:, tid] with 4 independent chains
            float a0 = 0.f, a1 = 0.f, a2 = 0.f, a3 = 0.f;
#pragma unroll
            for (int d = 0; d < DIM; d += 4) {
                a0 = fmaf(kv[d+0], S[d+0], a0);
                a1 = fmaf(kv[d+1], S[d+1], a1);
                a2 = fmaf(kv[d+2], S[d+2], a2);
                a3 = fmaf(kv[d+3], S[d+3], a3);
            }
            const float kS = (a0 + a1) + (a2 + a3);
            const float w  = sAlpha[j] * (sV[j * DIM + tid] - kS);
            // rank-1 update from registers (independent FMAs)
#pragma unroll
            for (int d = 0; d < DIM; ++d)
                S[d] = fmaf(kv[d], w, S[d]);
        }
        __syncthreads();   // protect sH/sK/sV/sAlpha before next block overwrites
    }

    // ---- Tail: q at last position (raw, per reference), readout, LN, classifier ----
    {
        float qn = 0.0f;
#pragma unroll
        for (int d = 0; d < DIM; ++d)
            qn = fmaf(Wq[tid * DIM + d], sH[15 * DIM + d], qn);
        sVec[tid] = qn;
    }
    __syncthreads();

    float last = 0.0f;
#pragma unroll
    for (int d = 0; d < DIM; ++d)
        last = fmaf(sVec[d], S[d], last);      // last[n] = q . S[:, n]
    __syncthreads();
    sVec[tid] = last;
    __syncthreads();

    float mu = 0.0f;
#pragma unroll
    for (int d = 0; d < DIM; ++d) mu += sVec[d];
    mu *= (1.0f / DIM);
    float var = 0.0f;
#pragma unroll
    for (int d = 0; d < DIM; ++d) {
        const float z = sVec[d] - mu;
        var = fmaf(z, z, var);
    }
    var *= (1.0f / DIM);
    const float rstd = rsqrtf(var + 1e-5f);
    const float ln   = (last - mu) * rstd * ln_g[tid] + ln_b[tid];
    sLN[tid] = ln;
    __syncthreads();

    if (tid < 10) {
        float o = bc[tid];
#pragma unroll
        for (int d = 0; d < DIM; ++d)
            o = fmaf(Wc[tid * DIM + d], sLN[d], o);
        out[b * 10 + tid] = o;
    }
}

extern "C" void kernel_launch(void* const* d_in, const int* in_sizes, int n_in,
                              void* d_out, int out_size, void* d_ws, size_t ws_size,
                              hipStream_t stream) {
    (void)in_sizes; (void)n_in; (void)out_size; (void)d_ws; (void)ws_size;
    efla_scan_kernel<<<256, 64, 0, stream>>>(
        (const float*)d_in[0],  // x
        (const float*)d_in[1],  // W_in
        (const float*)d_in[2],  // pos_emb
        (const float*)d_in[3],  // Wq
        (const float*)d_in[4],  // Wk
        (const float*)d_in[5],  // Wv
        (const float*)d_in[6],  // Wb
        (const float*)d_in[7],  // bb
        (const float*)d_in[8],  // ln_g
        (const float*)d_in[9],  // ln_b
        (const float*)d_in[10], // Wc
        (const float*)d_in[11], // bc
        (float*)d_out);
}
